// selfLSTM_3_20684562497778
// MI455X (gfx1250) — compile-verified, matrix operand order checked by reading, not run on hardware
//
#include <hip/hip_runtime.h>

// ---------------------------------------------------------------------------
// Types for CDNA5 WMMA (wave32): D(16x16 f32) = A(16x32 f16) * B(32x16 f16) + C
// ---------------------------------------------------------------------------
typedef _Float16 v16h __attribute__((ext_vector_type(16)));
typedef _Float16 v8h  __attribute__((ext_vector_type(8)));
typedef float    v8f  __attribute__((ext_vector_type(8)));

#define Lseq 128
#define Bb   16
#define Kk   512
#define Epad 128
#define NG   2048   // 4*K gate width

__device__ __forceinline__ v8f wmma16(v16h a, v16h b, v8f c) {
  return __builtin_amdgcn_wmma_f32_16x16x32_f16(false, a, false, b, (short)0, c,
                                                false, false);
}

// Load a 16x32 f16 fragment from a row-major matrix: element (r,k) at
// base[r*stride + k].  Works for both A (r=m) and B (r=n) operands.
// ISA layout: lanes 0-15 hold r=lane, k in {khi..khi+7, khi+16..khi+23} with
// khi=0; lanes 16-31 hold r=lane-16 with khi=8.
__device__ __forceinline__ v16h load_frag(const _Float16* base, int stride) {
  const int lane = threadIdx.x & 31;
  const int r    = lane & 15;
  const int khi  = (lane >> 4) << 3;            // 0 or 8
  const _Float16* p = base + r * stride + khi;  // 16B aligned (khi mult of 8)
  v8h lo = *(const v8h*)(p);
  v8h hi = *(const v8h*)(p + 16);
  v16h out;
#pragma unroll
  for (int i = 0; i < 8; ++i) { out[i] = lo[i]; out[i + 8] = hi[i]; }
  return out;
}

// Store 16x16 f32 accumulator: VGPR r -> row r + (lane>=16 ? 8:0), col lane&15
__device__ __forceinline__ void store_acc(float* base, int stride, v8f acc) {
  const int lane = threadIdx.x & 31;
  const int col  = lane & 15;
  const int rb   = (lane >> 4) << 3;
#pragma unroll
  for (int r = 0; r < 8; ++r) base[(rb + r) * stride + col] = acc[r];
}

__device__ __forceinline__ v8f vzero8() {
  v8f z;
#pragma unroll
  for (int q = 0; q < 8; ++q) z[q] = 0.0f;
  return z;
}

__device__ __forceinline__ float sigmf(float x) { return 1.0f / (1.0f + __expf(-x)); }

// ---------------------------------------------------------------------------
// Prep kernels
// ---------------------------------------------------------------------------
__global__ void k_convert_pad(const float* __restrict__ in, _Float16* __restrict__ out,
                              int rows, int cols, int colsPad) {
  int i = blockIdx.x * blockDim.x + threadIdx.x;
  if (i >= rows * colsPad) return;
  int r = i / colsPad, c = i - r * colsPad;
  out[i] = (_Float16)(c < cols ? in[r * cols + c] : 0.0f);
}

__global__ void k_gather_emb(const int* __restrict__ idx, const float* __restrict__ emb,
                             _Float16* __restrict__ out) {
  int i = blockIdx.x * blockDim.x + threadIdx.x;   // over L*B*Epad
  if (i >= Lseq * Bb * Epad) return;
  int rb = i >> 7, c = i & (Epad - 1);
  int row = idx[rb];
  out[i] = (_Float16)(c < 100 ? emb[row * 100 + c] : 0.0f);
}

__global__ void k_bias_add(const float* __restrict__ a, const float* __restrict__ b,
                           float* __restrict__ o, int n) {
  int i = blockIdx.x * blockDim.x + threadIdx.x;
  if (i < n) o[i] = a[i] + b[i];
}

// ---------------------------------------------------------------------------
// LSTM: one block = one LSTM, 8 waves, loops over 128 timesteps internally.
// g = x@Wih.T + h@Whh.T + bias ; gate order i,f,g,o ; h,c resident in LDS.
// ---------------------------------------------------------------------------
struct LstmArgs {
  const _Float16* x;    // (L, 16, 128) f16, padded embeddings
  const _Float16* Wih;  // (2048, 128) f16 row-major (n,k)
  const _Float16* Whh;  // (2048, 512) f16 row-major (n,k)
  const float*    bias; // (2048) f32 = bih + bhh
  const float*    h0;   // (16,512) f32 or nullptr -> zeros
  const float*    c0;   // (16,512) f32 or nullptr -> zeros
  _Float16*       Y;    // (L, 16, 512) f16 outputs (all hidden states)
  float*          hT;   // (16,512) f32 final h
  float*          cT;   // (16,512) f32 final c
};

__global__ void k_lstm(LstmArgs a0, LstmArgs a1) {
  LstmArgs A = (blockIdx.x == 0) ? a0 : a1;
  __shared__ __align__(16) _Float16 sh_h[16 * Kk];   // 16 KB
  __shared__ __align__(16) _Float16 sh_x[16 * Epad]; //  4 KB
  __shared__ float sh_c[16 * Kk];                    // 32 KB
  const int tid = threadIdx.x, wave = tid >> 5, lane = tid & 31;

  for (int i = tid; i < 16 * Kk; i += 256) {
    sh_c[i] = A.c0 ? A.c0[i] : 0.0f;
    sh_h[i] = (_Float16)(A.h0 ? A.h0[i] : 0.0f);
  }
  __syncthreads();

  for (int t = 0; t < Lseq; ++t) {
    // stage x_t (16x128 f16): 256 threads x 8 halves
    ((v8h*)sh_x)[tid] = ((const v8h*)(A.x + (size_t)t * 16 * Epad))[tid];
    __syncthreads();

    _Float16 hreg[4][8];
#pragma unroll
    for (int ct = 0; ct < 4; ++ct) {
      const int ncol = wave * 64 + ct * 16;  // column tile within K=512
      v8f acc[4];
#pragma unroll
      for (int g = 0; g < 4; ++g) acc[g] = vzero8();
      // x @ Wih.T  (K = 128 padded)
#pragma unroll
      for (int kk = 0; kk < 4; ++kk) {
        v16h a = load_frag(sh_x + kk * 32, Epad);
#pragma unroll
        for (int g = 0; g < 4; ++g) {
          v16h bf = load_frag(A.Wih + (size_t)(g * Kk + ncol) * Epad + kk * 32, Epad);
          acc[g] = wmma16(a, bf, acc[g]);
        }
      }
      // h @ Whh.T  (K = 512)
      for (int kk = 0; kk < 16; ++kk) {
        v16h a = load_frag(sh_h + kk * 32, Kk);
#pragma unroll
        for (int g = 0; g < 4; ++g) {
          v16h bf = load_frag(A.Whh + (size_t)(g * Kk + ncol) * Kk + kk * 32, Kk);
          acc[g] = wmma16(a, bf, acc[g]);
        }
      }
      // gates + cell update (each (row,col) owned by exactly one lane)
      const int col = ncol + (lane & 15);
      const int rb  = (lane >> 4) << 3;
      const float bi = A.bias[col], bf2 = A.bias[Kk + col];
      const float bg = A.bias[2 * Kk + col], bo = A.bias[3 * Kk + col];
#pragma unroll
      for (int r = 0; r < 8; ++r) {
        const int row = rb + r;
        float iv = sigmf(acc[0][r] + bi);
        float fv = sigmf(acc[1][r] + bf2);
        float gv = tanhf(acc[2][r] + bg);
        float ov = sigmf(acc[3][r] + bo);
        float cc = fv * sh_c[row * Kk + col] + iv * gv;
        sh_c[row * Kk + col] = cc;
        hreg[ct][r] = (_Float16)(ov * tanhf(cc));
      }
    }
    __syncthreads();  // all waves finished reading old sh_h / sh_x
#pragma unroll
    for (int ct = 0; ct < 4; ++ct) {
      const int col = wave * 64 + ct * 16 + (lane & 15);
      const int rb  = (lane >> 4) << 3;
#pragma unroll
      for (int r = 0; r < 8; ++r) {
        const int row = rb + r;
        sh_h[row * Kk + col] = hreg[ct][r];
        A.Y[(size_t)t * 16 * Kk + row * Kk + col] = hreg[ct][r];
      }
    }
    __syncthreads();
  }
  for (int i = tid; i < 16 * Kk; i += 256) {
    A.hT[i] = (float)sh_h[i];
    A.cT[i] = sh_c[i];
  }
}

// ---------------------------------------------------------------------------
// Y_Wy[b,j,l] = sum_k Wy[j,k] * Y[l,b,k]    (per-batch 512x128 GEMM, K=512)
// grid = 32 blocks: branch = bid>>4, b = bid&15
// ---------------------------------------------------------------------------
__global__ void k_ywy(const _Float16* Wy1, const _Float16* Wy2,
                      const _Float16* Y1, const _Float16* Y2,
                      float* YWy1, float* YWy2) {
  const int br = blockIdx.x >> 4;
  const int b  = blockIdx.x & 15;
  const _Float16* Wy = br ? Wy2 : Wy1;
  const _Float16* Y  = br ? Y2 : Y1;
  float* YW = br ? YWy2 : YWy1;
  const int wave = threadIdx.x >> 5;
  for (int tt = wave; tt < 256; tt += 8) {       // 32 j-tiles x 8 l-tiles
    const int j0 = (tt >> 3) << 4;
    const int l0 = (tt & 7) << 4;
    v8f acc = vzero8();
    for (int kk = 0; kk < 16; ++kk) {
      v16h a  = load_frag(Wy + (size_t)j0 * Kk + kk * 32, Kk);
      v16h bf = load_frag(Y + ((size_t)l0 * 16 + b) * Kk + kk * 32, 16 * Kk);
      acc = wmma16(a, bf, acc);
    }
    store_acc(YW + ((size_t)b * Kk + j0) * Lseq + l0, Lseq, acc);
  }
}

// ---------------------------------------------------------------------------
// Attention scan: r_{t} = Y^T softmax_l( wt . tanh(hW + YWy) )
// one block per branch; r resident in LDS.
// ---------------------------------------------------------------------------
struct AttArgs {
  const _Float16* Yr;    // (L,16,512) driving states
  const _Float16* Yatt;  // (L,16,512) attended states
  const float*    YWy;   // (16,512,128) precomputed
  const _Float16* Whf;   // (512,512) f16 (n,k)
  const _Float16* Wrf;   // (512,512) f16 (n,k)
  const float*    wt;    // (512)
  float*          rout;  // (16,512) final r
};

__global__ void k_attend(AttArgs a0, AttArgs a1) {
  AttArgs A = (blockIdx.x == 0) ? a0 : a1;
  __shared__ __align__(16) _Float16 sh_r[16 * Kk];  // 16 KB
  __shared__ float sh_hW[16 * Kk];                  // 32 KB
  __shared__ float sh_s[16 * Lseq];                 //  8 KB
  __shared__ float sh_wt[Kk];                       //  2 KB
  const int tid = threadIdx.x, wave = tid >> 5;

  for (int i = tid; i < 16 * Kk; i += 256) sh_r[i] = (_Float16)0.0f;
  for (int i = tid; i < Kk; i += 256) sh_wt[i] = A.wt[i];
  __syncthreads();

  for (int t = 0; t < Lseq; ++t) {
    const _Float16* ht = A.Yr + (size_t)t * 16 * Kk;
    // hW = h_t @ Wh.T + r @ Wr.T   (16x512, two K=512 reductions)
#pragma unroll
    for (int ct = 0; ct < 4; ++ct) {
      const int ncol = wave * 64 + ct * 16;
      v8f acc = vzero8();
      for (int kk = 0; kk < 16; ++kk) {
        v16h a  = load_frag(ht + kk * 32, Kk);
        v16h bf = load_frag(A.Whf + (size_t)ncol * Kk + kk * 32, Kk);
        acc = wmma16(a, bf, acc);
      }
      for (int kk = 0; kk < 16; ++kk) {
        v16h a  = load_frag(sh_r + kk * 32, Kk);
        v16h bf = load_frag(A.Wrf + (size_t)ncol * Kk + kk * 32, Kk);
        acc = wmma16(a, bf, acc);
      }
      store_acc(sh_hW + ncol, Kk, acc);
    }
    __syncthreads();
    // scores s[b,l] = sum_k tanh(hW[b,k] + YWy[b,k,l]) * wt[k]
    for (int p = tid; p < 16 * Lseq; p += 256) {
      const int b = p >> 7, l = p & (Lseq - 1);
      const float* hw = sh_hW + b * Kk;
      const float* yw = A.YWy + (size_t)b * Kk * Lseq + l;
      float s = 0.0f;
      for (int k = 0; k < Kk; ++k)
        s += tanhf(hw[k] + yw[(size_t)k * Lseq]) * sh_wt[k];
      sh_s[p] = s;
    }
    __syncthreads();
    if (tid < 16) {  // softmax over l per batch row
      float m = -1e30f;
      for (int l = 0; l < Lseq; ++l) m = fmaxf(m, sh_s[tid * Lseq + l]);
      float sum = 0.0f;
      for (int l = 0; l < Lseq; ++l) {
        float e = __expf(sh_s[tid * Lseq + l] - m);
        sh_s[tid * Lseq + l] = e;
        sum += e;
      }
      float inv = 1.0f / sum;
      for (int l = 0; l < Lseq; ++l) sh_s[tid * Lseq + l] *= inv;
    }
    __syncthreads();
    // r[b,k] = sum_l Y[l,b,k] * alpha[b,l]
    for (int o = tid; o < 16 * Kk; o += 256) {
      const int b = o >> 9, k = o & (Kk - 1);
      float v = 0.0f;
      for (int l = 0; l < Lseq; ++l)
        v += (float)A.Yatt[((size_t)l * 16 + b) * Kk + k] * sh_s[b * Lseq + l];
      sh_r[o] = (_Float16)v;
      if (t == Lseq - 1) A.rout[o] = v;
    }
    __syncthreads();
  }
}

// ---------------------------------------------------------------------------
// out12[b, j] = tanh( r@Wp.T + hn@Wx.T )  (fp32, tiny)
// ---------------------------------------------------------------------------
__global__ void k_proj(const float* r1, const float* hn1, const float* Wx1, const float* Wp1,
                       const float* r2, const float* hn2, const float* Wx2, const float* Wp2,
                       float* out12) {
  int i = blockIdx.x * blockDim.x + threadIdx.x;
  if (i >= 16 * 1024) return;
  int b = i >> 10, j = i & 1023;
  const float *r, *hn, *Wx, *Wp;
  int jj;
  if (j < 512) { r = r1; hn = hn1; Wx = Wx1; Wp = Wp1; jj = j; }
  else         { r = r2; hn = hn2; Wx = Wx2; Wp = Wp2; jj = j - 512; }
  float s = 0.0f;
  for (int k = 0; k < Kk; ++k)
    s += r[b * Kk + k] * Wp[jj * Kk + k] + hn[b * Kk + k] * Wx[jj * Kk + k];
  out12[i] = tanhf(s);
}

__global__ void k_logits(const float* out12, const float* Wout, const float* bout,
                         float* out) {
  int tid = threadIdx.x;
  if (tid >= 48) return;
  int b = tid / 3, o = tid - b * 3;
  float s = bout[o];
  for (int j = 0; j < 1024; ++j) s += out12[b * 1024 + j] * Wout[o * 1024 + j];
  out[b * 3 + o] = s;
}

// ---------------------------------------------------------------------------
// Host driver
// ---------------------------------------------------------------------------
extern "C" void kernel_launch(void* const* d_in, const int* in_sizes, int n_in,
                              void* d_out, int out_size, void* d_ws, size_t ws_size,
                              hipStream_t stream) {
  (void)in_sizes; (void)n_in; (void)out_size; (void)ws_size;
  // setup_inputs() dict order:
  const int*   xs1 = (const int*)d_in[0];
  const int*   xs2 = (const int*)d_in[1];
  const float* emb = (const float*)d_in[2];
  // LSTM param blocks: l1=3, r1=7, l2=11, r2=15 (each Wih,Whh,bih,bhh)
  const int lstm_base[4] = {3, 7, 11, 15};  // order: l1, r1, l2, r2
  // attention: Wy1=19 Wh1=20 Wx1=21 Wp1=22 Wr1=23 Wt1=24
  //            Wy2=25 Wh2=26 Wx2=27 Wp2=28 Wr2=29 Wt2=30, Wout=31, bout=32
  const float* Wy1 = (const float*)d_in[19];
  const float* Wh1 = (const float*)d_in[20];
  const float* Wx1 = (const float*)d_in[21];
  const float* Wp1 = (const float*)d_in[22];
  const float* Wr1 = (const float*)d_in[23];
  const float* wt1 = (const float*)d_in[24];
  const float* Wy2 = (const float*)d_in[25];
  const float* Wh2 = (const float*)d_in[26];
  const float* Wx2 = (const float*)d_in[27];
  const float* Wp2 = (const float*)d_in[28];
  const float* Wr2 = (const float*)d_in[29];
  const float* wt2 = (const float*)d_in[30];
  const float* Wout = (const float*)d_in[31];
  const float* bout = (const float*)d_in[32];

  // ---- workspace carve-out (256B aligned) ----
  char* W = (char*)d_ws;
  size_t off = 0;
  auto carve = [&](size_t bytes) {
    void* p = W + off;
    off = (off + bytes + 255) & ~(size_t)255;
    return p;
  };
  _Float16* e1h = (_Float16*)carve((size_t)Lseq * Bb * Epad * 2);
  _Float16* e2h = (_Float16*)carve((size_t)Lseq * Bb * Epad * 2);
  _Float16* WihH[4]; _Float16* WhhH[4]; float* biasF[4];
  for (int i = 0; i < 4; ++i) {
    WihH[i]  = (_Float16*)carve((size_t)NG * Epad * 2);
    WhhH[i]  = (_Float16*)carve((size_t)NG * Kk * 2);
    biasF[i] = (float*)carve((size_t)NG * 4);
  }
  _Float16* Wy1h = (_Float16*)carve((size_t)Kk * Kk * 2);
  _Float16* Wh1h = (_Float16*)carve((size_t)Kk * Kk * 2);
  _Float16* Wr1h = (_Float16*)carve((size_t)Kk * Kk * 2);
  _Float16* Wy2h = (_Float16*)carve((size_t)Kk * Kk * 2);
  _Float16* Wh2h = (_Float16*)carve((size_t)Kk * Kk * 2);
  _Float16* Wr2h = (_Float16*)carve((size_t)Kk * Kk * 2);
  _Float16* Y1  = (_Float16*)carve((size_t)Lseq * Bb * Kk * 2);
  _Float16* Yr1 = (_Float16*)carve((size_t)Lseq * Bb * Kk * 2);
  _Float16* Y2  = (_Float16*)carve((size_t)Lseq * Bb * Kk * 2);
  _Float16* Yr2 = (_Float16*)carve((size_t)Lseq * Bb * Kk * 2);
  float* hl1 = (float*)carve((size_t)Bb * Kk * 4);
  float* cl1 = (float*)carve((size_t)Bb * Kk * 4);
  float* hl2 = (float*)carve((size_t)Bb * Kk * 4);
  float* cl2 = (float*)carve((size_t)Bb * Kk * 4);
  float* hn1 = (float*)carve((size_t)Bb * Kk * 4);
  float* cn1 = (float*)carve((size_t)Bb * Kk * 4);
  float* hn2 = (float*)carve((size_t)Bb * Kk * 4);
  float* cn2 = (float*)carve((size_t)Bb * Kk * 4);
  float* YWy1 = (float*)carve((size_t)Bb * Kk * Lseq * 4);
  float* YWy2 = (float*)carve((size_t)Bb * Kk * Lseq * 4);
  float* r1f = (float*)carve((size_t)Bb * Kk * 4);
  float* r2f = (float*)carve((size_t)Bb * Kk * 4);
  float* out12 = (float*)carve((size_t)Bb * 1024 * 4);

  // ---- phase 0: convert weights -> f16, gather embeddings, fold biases ----
  auto cvt = [&](const float* in, _Float16* outp, int rows, int cols, int colsPad) {
    int total = rows * colsPad;
    k_convert_pad<<<(total + 255) / 256, 256, 0, stream>>>(in, outp, rows, cols, colsPad);
  };
  {
    int total = Lseq * Bb * Epad;
    k_gather_emb<<<(total + 255) / 256, 256, 0, stream>>>(xs1, emb, e1h);
    k_gather_emb<<<(total + 255) / 256, 256, 0, stream>>>(xs2, emb, e2h);
  }
  for (int i = 0; i < 4; ++i) {
    int bi = lstm_base[i];
    cvt((const float*)d_in[bi + 0], WihH[i], NG, 100, Epad);
    cvt((const float*)d_in[bi + 1], WhhH[i], NG, Kk, Kk);
    k_bias_add<<<(NG + 255) / 256, 256, 0, stream>>>(
        (const float*)d_in[bi + 2], (const float*)d_in[bi + 3], biasF[i], NG);
  }
  cvt(Wy1, Wy1h, Kk, Kk, Kk);  cvt(Wh1, Wh1h, Kk, Kk, Kk);  cvt(Wr1, Wr1h, Kk, Kk, Kk);
  cvt(Wy2, Wy2h, Kk, Kk, Kk);  cvt(Wh2, Wh2h, Kk, Kk, Kk);  cvt(Wr2, Wr2h, Kk, Kk, Kk);

  // ---- phase 1: l1 over e1 and l2 over e2 (independent -> 2 blocks) ----
  LstmArgs al1 = {e1h, WihH[0], WhhH[0], biasF[0], nullptr, nullptr, Y1, hl1, cl1};
  LstmArgs al2 = {e2h, WihH[2], WhhH[2], biasF[2], nullptr, nullptr, Y2, hl2, cl2};
  k_lstm<<<2, 256, 0, stream>>>(al1, al2);

  // ---- phase 2: r1 over e2 (init hl1/cl1), r2 over e1 (init hl2/cl2) ----
  LstmArgs ar1 = {e2h, WihH[1], WhhH[1], biasF[1], hl1, cl1, Yr1, hn1, cn1};
  LstmArgs ar2 = {e1h, WihH[3], WhhH[3], biasF[3], hl2, cl2, Yr2, hn2, cn2};
  k_lstm<<<2, 256, 0, stream>>>(ar1, ar2);

  // ---- phase 3: precompute Y_Wy for both branches ----
  k_ywy<<<32, 256, 0, stream>>>(Wy1h, Wy2h, Y1, Y2, YWy1, YWy2);

  // ---- phase 4: attention scans ----
  AttArgs at1 = {Yr1, Y1, YWy1, Wh1h, Wr1h, wt1, r1f};
  AttArgs at2 = {Yr2, Y2, YWy2, Wh2h, Wr2h, wt2, r2f};
  k_attend<<<2, 256, 0, stream>>>(at1, at2);

  // ---- phase 5: projections + logits (fp32, tiny) ----
  k_proj<<<64, 256, 0, stream>>>(r1f, hn1, Wx1, Wp1, r2f, hn2, Wx2, Wp2, out12);
  k_logits<<<1, 64, 0, stream>>>(out12, Wout, bout, (float*)d_out);
}
